// Encoder_4904852652842
// MI455X (gfx1250) — compile-verified, matrix operand order checked by reading, not run on hardware
//
#include <hip/hip_runtime.h>

// ---------------------------------------------------------------------------
// MI455X (gfx1250) encoder forward.
// All GEMMs via v_wmma_f32_16x16x32_bf16 (wave32, fp32 accumulate).
// GEMM inner loop: K-chunk 64, double-buffered LDS with register staging
// (global loads for stage s+1 issued before stage s's WMMAs; one barrier per
// stage) + global_prefetch of stage s+2 to warm L2. LayerNorm / softmax /
// bias / residual in fp32. Attention maps written once to d_out (536 MB fp32
// = the HBM roofline floor) and softmaxed in place, with a bf16 copy kept
// for the AV GEMM. Workspace use: ~210 MB.
// ---------------------------------------------------------------------------

typedef __bf16 bf16x16 __attribute__((ext_vector_type(16)));
typedef float  floatx8 __attribute__((ext_vector_type(8)));

#define TILE_M 64
#define TILE_N 64
#define TILE_K 64
#define LDSTR  68   // LDS row stride in halves (136B rows: even, 8B-aligned)
#define LBUF   (64 * LDSTR)

__device__ __forceinline__ unsigned short f2bf(float f) {
    unsigned u = __float_as_uint(f);
    unsigned r = u + 0x7FFFu + ((u >> 16) & 1u);   // round-to-nearest-even
    return (unsigned short)(r >> 16);
}

// ---------------------------------------------------------------------------
// Batched bf16 GEMM:  C[bz] = epilogue( A[bz] (MxK) * B[bz] (KxN) )
// OUTK: 0 = fp32 C, 1 = bf16 C, 2 = bf16 C transposed (C[n*ldc + m])
// Block = 128 threads = 4 waves; each wave owns a 32x32 output quadrant
// (2x2 WMMA 16x16 tiles). Requires M%64==0, N%64==0, K%64==0 (true here).
// ---------------------------------------------------------------------------
template <int OUTK, bool BIAS, bool RELU, bool RESID>
__global__ __launch_bounds__(128)
void gemm_bf16_kernel(const unsigned short* __restrict__ A,
                      const unsigned short* __restrict__ B,
                      void* __restrict__ Cv,
                      const float* __restrict__ bias,
                      const float* __restrict__ resid,
                      int M, int N, int K, int lda, int ldb, int ldc,
                      float scale,
                      long long bsA, long long bsB, long long bsC) {
    __shared__ unsigned short As[2 * LBUF];   // A tiles, row-major [64][TILE_K]
    __shared__ unsigned short Bt[2 * LBUF];   // B tiles, stored transposed [64 n][TILE_K]

    const int tid  = threadIdx.x;
    const int lane = tid & 31;
    const int wave = tid >> 5;
    const int wrow = (wave >> 1) * 32;
    const int wcol = (wave & 1) * 32;

    const int m0 = blockIdx.y * TILE_M;
    const int n0 = blockIdx.x * TILE_N;
    const long long bz = blockIdx.z;

    A += bz * bsA;
    B += bz * bsB;

    floatx8 acc00 = {0.f,0.f,0.f,0.f,0.f,0.f,0.f,0.f};
    floatx8 acc01 = acc00, acc10 = acc00, acc11 = acc00;

    const int lm = lane & 15;
    const int kb = (lane >> 4) * 8;
    union FragU { bf16x16 v; unsigned u[8]; };

    // ---- prologue: stage 0 directly into buffer 0 ----
    #pragma unroll
    for (int i = 0; i < 8; ++i) {
        int g   = (i * 128 + tid) * 4;
        int row = g >> 6;
        int col = g & 63;
        uint2 av = *(const uint2*)(A + (size_t)(m0 + row) * lda + col);
        *(uint2*)(&As[row * LDSTR + col]) = av;
        uint2 bv = *(const uint2*)(B + (size_t)row * ldb + n0 + col);
        const unsigned short* sp = (const unsigned short*)&bv;
        Bt[(col + 0) * LDSTR + row] = sp[0];
        Bt[(col + 1) * LDSTR + row] = sp[1];
        Bt[(col + 2) * LDSTR + row] = sp[2];
        Bt[(col + 3) * LDSTR + row] = sp[3];
    }
    __syncthreads();

    const int nk = K / TILE_K;
    int buf = 0;
    for (int s = 0; s < nk; ++s) {
        const int k0   = s * TILE_K;
        const bool more = (s + 1 < nk);

        // --- issue next stage's global loads into registers (overlaps WMMA) ---
        uint2 ra[8], rb[8];
        if (more) {
            const int kn = k0 + TILE_K;
            #pragma unroll
            for (int i = 0; i < 8; ++i) {
                int g = (i * 128 + tid) * 4;
                ra[i] = *(const uint2*)(A + (size_t)(m0 + (g >> 6)) * lda + kn + (g & 63));
                rb[i] = *(const uint2*)(B + (size_t)(kn + (g >> 6)) * ldb + n0 + (g & 63));
            }
            // warm L2 two stages ahead (global_prefetch_b8)
            if (k0 + 2 * TILE_K < K) {
                __builtin_prefetch(A + (size_t)(m0 + (tid & 63)) * lda + k0 + 2 * TILE_K, 0, 1);
                __builtin_prefetch(B + (size_t)(k0 + 2 * TILE_K + (tid >> 1)) * ldb + n0, 0, 1);
            }
        }

        // --- compute: 2 K-subchunks x 4 WMMA from buffer `buf` ---
        const unsigned short* Ab = As + buf * LBUF;
        const unsigned short* Bb = Bt + buf * LBUF;
        #pragma unroll
        for (int c = 0; c < 2; ++c) {
            FragU a0, a1, b0, b1;
            #pragma unroll
            for (int j = 0; j < 8; ++j) {
                // ISA 16-bit A layout: lane=row, VGPR j ->
                // K = c*32 + (j>=4 ? 16 : 0) + (lane>=16 ? 8 : 0) + 2*(j&3)
                int kk = c * 32 + ((j & 4) ? 16 : 0) + kb + 2 * (j & 3);
                a0.u[j] = *(const unsigned*)(&Ab[(wrow      + lm) * LDSTR + kk]);
                a1.u[j] = *(const unsigned*)(&Ab[(wrow + 16 + lm) * LDSTR + kk]);
                b0.u[j] = *(const unsigned*)(&Bb[(wcol      + lm) * LDSTR + kk]);
                b1.u[j] = *(const unsigned*)(&Bb[(wcol + 16 + lm) * LDSTR + kk]);
            }
            acc00 = __builtin_amdgcn_wmma_f32_16x16x32_bf16(false, a0.v, false, b0.v, (short)0, acc00, false, false);
            acc01 = __builtin_amdgcn_wmma_f32_16x16x32_bf16(false, a0.v, false, b1.v, (short)0, acc01, false, false);
            acc10 = __builtin_amdgcn_wmma_f32_16x16x32_bf16(false, a1.v, false, b0.v, (short)0, acc10, false, false);
            acc11 = __builtin_amdgcn_wmma_f32_16x16x32_bf16(false, a1.v, false, b1.v, (short)0, acc11, false, false);
        }

        // --- drain staged registers into the other buffer ---
        if (more) {
            unsigned short* An = As + (buf ^ 1) * LBUF;
            unsigned short* Bn = Bt + (buf ^ 1) * LBUF;
            #pragma unroll
            for (int i = 0; i < 8; ++i) {
                int g  = (i * 128 + tid) * 4;
                int rw = g >> 6;
                int cl = g & 63;
                *(uint2*)(&An[rw * LDSTR + cl]) = ra[i];
                const unsigned short* sp = (const unsigned short*)&rb[i];
                Bn[(cl + 0) * LDSTR + rw] = sp[0];
                Bn[(cl + 1) * LDSTR + rw] = sp[1];
                Bn[(cl + 2) * LDSTR + rw] = sp[2];
                Bn[(cl + 3) * LDSTR + rw] = sp[3];
            }
        }
        __syncthreads();
        buf ^= 1;
    }

    // --- epilogue: C layout = lane holds col (lane&15), VGPR r -> row r+(lane>=16?8:0) ---
    float*          Cf = (float*)Cv;
    unsigned short* Cb = (unsigned short*)Cv;
    const int rhalf = (lane >> 4) * 8;

    auto emit = [&](floatx8 acc, int dm, int dn) {
        const int col = n0 + wcol + dn + lm;
        #pragma unroll
        for (int r = 0; r < 8; ++r) {
            const int row = m0 + wrow + dm + rhalf + r;
            float c = acc[r] * scale;
            if (BIAS)  c += bias[col];
            if (RELU)  c = fmaxf(c, 0.0f);
            if (RESID) c += resid[(size_t)row * ldc + col];
            if (OUTK == 0)      Cf[bz * bsC + (size_t)row * ldc + col] = c;
            else if (OUTK == 1) Cb[bz * bsC + (size_t)row * ldc + col] = f2bf(c);
            else                Cb[bz * bsC + (size_t)col * ldc + row] = f2bf(c);
        }
    };
    emit(acc00, 0, 0);  emit(acc01, 0, 16);
    emit(acc10, 16, 0); emit(acc11, 16, 16);
}

// ---------------------------------------------------------------------------
__global__ __launch_bounds__(256)
void embed_kernel(const int* __restrict__ seq, const float* __restrict__ tok,
                  const float* __restrict__ pos, float* __restrict__ x, int E) {
    int i = blockIdx.x * 256 + threadIdx.x;
    int s = i / E, e = i % E;
    x[i] = tok[(size_t)seq[s] * E + e] + pos[i];
}

__global__ __launch_bounds__(256)
void convert_bf16_kernel(const float* __restrict__ src, unsigned short* __restrict__ dst) {
    size_t i = (size_t)blockIdx.x * 256 + threadIdx.x;
    dst[i] = f2bf(src[i]);
}

// One row (E=1024) per block; writes fp32 + bf16 normalized output.
__global__ __launch_bounds__(256)
void layernorm_kernel(const float* __restrict__ x, const float* __restrict__ g,
                      const float* __restrict__ b, float* __restrict__ y,
                      unsigned short* __restrict__ ybf, int E) {
    __shared__ float red[256];
    const int tid = threadIdx.x;
    const float* xr = x + (size_t)blockIdx.x * E;
    float s = 0.f;
    for (int e = tid; e < E; e += 256) s += xr[e];
    red[tid] = s; __syncthreads();
    for (int o = 128; o > 0; o >>= 1) { if (tid < o) red[tid] += red[tid + o]; __syncthreads(); }
    const float mu = red[0] / E; __syncthreads();
    float v = 0.f;
    for (int e = tid; e < E; e += 256) { float d = xr[e] - mu; v += d * d; }
    red[tid] = v; __syncthreads();
    for (int o = 128; o > 0; o >>= 1) { if (tid < o) red[tid] += red[tid + o]; __syncthreads(); }
    const float inv = rsqrtf(red[0] / E + 1e-5f);
    float* yr = y + (size_t)blockIdx.x * E;
    unsigned short* ybr = ybf + (size_t)blockIdx.x * E;
    for (int e = tid; e < E; e += 256) {
        float o = (xr[e] - mu) * inv * g[e] + b[e];
        yr[e] = o; ybr[e] = f2bf(o);
    }
}

// One row of 2048 scores per block; fp32 softmax in place + bf16 copy.
__global__ __launch_bounds__(256)
void softmax_kernel(float* __restrict__ att, unsigned short* __restrict__ attbf) {
    __shared__ float red[256];
    const int tid = threadIdx.x;
    float* a = att + (size_t)blockIdx.x * 2048;
    unsigned short* ab = attbf + (size_t)blockIdx.x * 2048;
    float vals[8];
    float m = -3.0e38f;
    #pragma unroll
    for (int i = 0; i < 8; ++i) { vals[i] = a[tid + i * 256]; m = fmaxf(m, vals[i]); }
    red[tid] = m; __syncthreads();
    for (int o = 128; o > 0; o >>= 1) { if (tid < o) red[tid] = fmaxf(red[tid], red[tid + o]); __syncthreads(); }
    m = red[0]; __syncthreads();
    float s = 0.f;
    #pragma unroll
    for (int i = 0; i < 8; ++i) { vals[i] = __expf(vals[i] - m); s += vals[i]; }
    red[tid] = s; __syncthreads();
    for (int o = 128; o > 0; o >>= 1) { if (tid < o) red[tid] += red[tid + o]; __syncthreads(); }
    const float inv = 1.0f / red[0];
    #pragma unroll
    for (int i = 0; i < 8; ++i) {
        float p = vals[i] * inv;
        a[tid + i * 256] = p;
        ab[tid + i * 256] = f2bf(p);
    }
}

// ---------------------------------------------------------------------------
extern "C" void kernel_launch(void* const* d_in, const int* in_sizes, int n_in,
                              void* d_out, int out_size, void* d_ws, size_t ws_size,
                              hipStream_t stream) {
    constexpr int S = 2048, E = 1024, H = 16, HS = 64, Lnum = 2, FF = 4096;
    typedef long long ll;

    const int*   seq  = (const int*)d_in[0];
    const float* tok  = (const float*)d_in[1];
    const float* pos  = (const float*)d_in[2];
    const float* Wq   = (const float*)d_in[3];   // [L,H,E,HS]
    const float* Wk   = (const float*)d_in[4];
    const float* Wv   = (const float*)d_in[5];
    const float* Wr   = (const float*)d_in[6];   // [L,E,E]
    const float* br   = (const float*)d_in[7];   // [L,E]
    const float* W1   = (const float*)d_in[8];   // [L,E,FF]
    const float* b1   = (const float*)d_in[9];   // [L,FF]
    const float* W2   = (const float*)d_in[10];  // [L,FF,E]
    const float* b2   = (const float*)d_in[11];  // [L,E]
    const float* ln_g = (const float*)d_in[12];  // [L,E]
    const float* ln_b = (const float*)d_in[13];

    float* out      = (float*)d_out;
    float* att_base = out + (size_t)S * E;       // [L,H,S,S]

    // ---- workspace carve-up (256B aligned) ----
    char* p = (char*)d_ws;
    auto alloc = [&](size_t bytes) { void* r = p; p += (bytes + 255) & ~(size_t)255; return r; };
    float*          xA    = (float*)alloc((size_t)S * E * 4);
    float*          xB    = (float*)alloc((size_t)S * E * 4);
    unsigned short* xn_bf = (unsigned short*)alloc((size_t)S * E * 2);
    unsigned short* q_bf  = (unsigned short*)alloc((size_t)H * S * HS * 2);
    unsigned short* kT_bf = (unsigned short*)alloc((size_t)H * HS * S * 2);
    unsigned short* v_bf  = (unsigned short*)alloc((size_t)H * S * HS * 2);
    unsigned short* cat_bf= (unsigned short*)alloc((size_t)S * E * 2);
    unsigned short* h1_bf = (unsigned short*)alloc((size_t)S * FF * 2);
    unsigned short* att_bf= (unsigned short*)alloc((size_t)H * S * S * 2);
    unsigned short* Wq_bf = (unsigned short*)alloc((size_t)H * E * HS * 2);
    unsigned short* Wk_bf = (unsigned short*)alloc((size_t)H * E * HS * 2);
    unsigned short* Wv_bf = (unsigned short*)alloc((size_t)H * E * HS * 2);
    unsigned short* Wr_bf = (unsigned short*)alloc((size_t)E * E * 2);
    unsigned short* W1_bf = (unsigned short*)alloc((size_t)E * FF * 2);
    unsigned short* W2_bf = (unsigned short*)alloc((size_t)FF * E * 2);

    embed_kernel<<<(S * E) / 256, 256, 0, stream>>>(seq, tok, pos, xA, E);

    for (int l = 0; l < Lnum; ++l) {
        // weight conversion fp32 -> bf16
        convert_bf16_kernel<<<(H * E * HS) / 256, 256, 0, stream>>>(Wq + (size_t)l * H * E * HS, Wq_bf);
        convert_bf16_kernel<<<(H * E * HS) / 256, 256, 0, stream>>>(Wk + (size_t)l * H * E * HS, Wk_bf);
        convert_bf16_kernel<<<(H * E * HS) / 256, 256, 0, stream>>>(Wv + (size_t)l * H * E * HS, Wv_bf);
        convert_bf16_kernel<<<(E * E)      / 256, 256, 0, stream>>>(Wr + (size_t)l * E * E, Wr_bf);
        convert_bf16_kernel<<<(E * FF)     / 256, 256, 0, stream>>>(W1 + (size_t)l * E * FF, W1_bf);
        convert_bf16_kernel<<<(FF * E)     / 256, 256, 0, stream>>>(W2 + (size_t)l * FF * E, W2_bf);

        // ln1: xB = LN(xA), xn_bf = bf16(xB)
        layernorm_kernel<<<S, 256, 0, stream>>>(xA, ln_g + (size_t)l * E, ln_b + (size_t)l * E, xB, xn_bf, E);

        // QKV per-head GEMMs, batched over heads (grid.z = H)
        dim3 gqkv(HS / 64, S / 64, H);
        gemm_bf16_kernel<1, false, false, false><<<gqkv, 128, 0, stream>>>(
            xn_bf, Wq_bf, q_bf, nullptr, nullptr, S, HS, E, E, HS, HS, 1.0f,
            0, (ll)E * HS, (ll)S * HS);
        gemm_bf16_kernel<2, false, false, false><<<gqkv, 128, 0, stream>>>(   // K stored transposed [H,HS,S]
            xn_bf, Wk_bf, kT_bf, nullptr, nullptr, S, HS, E, E, HS, S, 1.0f,
            0, (ll)E * HS, (ll)HS * S);
        gemm_bf16_kernel<1, false, false, false><<<gqkv, 128, 0, stream>>>(
            xn_bf, Wv_bf, v_bf, nullptr, nullptr, S, HS, E, E, HS, HS, 1.0f,
            0, (ll)E * HS, (ll)S * HS);

        // scores = (q @ k^T) / 8 -> fp32 straight into d_out attention slice
        float* att = att_base + (size_t)l * H * S * S;
        dim3 gsc(S / 64, S / 64, H);
        gemm_bf16_kernel<0, false, false, false><<<gsc, 128, 0, stream>>>(
            q_bf, kT_bf, att, nullptr, nullptr, S, S, HS, HS, S, S, 0.125f,
            (ll)S * HS, (ll)HS * S, (ll)S * S);

        // softmax in place (fp32) + bf16 copy for AV
        softmax_kernel<<<H * S, 256, 0, stream>>>(att, att_bf);

        // AV: cat[:, h*HS:(h+1)*HS] = att_h @ v_h  (bf16 out, batched heads)
        dim3 gav(HS / 64, S / 64, H);
        gemm_bf16_kernel<1, false, false, false><<<gav, 128, 0, stream>>>(
            att_bf, v_bf, cat_bf, nullptr, nullptr, S, HS, S, S, HS, E, 1.0f,
            (ll)S * S, (ll)S * HS, (ll)HS);

        // out-proj + bias + residual(xB) -> xA (fp32)
        dim3 gpr(E / 64, S / 64, 1);
        gemm_bf16_kernel<0, true, false, true><<<gpr, 128, 0, stream>>>(
            cat_bf, Wr_bf, xA, br + (size_t)l * E, xB, S, E, E, E, E, E, 1.0f, 0, 0, 0);

        // ln2: xB = LN(xA), xn_bf
        layernorm_kernel<<<S, 256, 0, stream>>>(xA, ln_g + (size_t)l * E, ln_b + (size_t)l * E, xB, xn_bf, E);

        // MLP: h1 = relu(xn @ W1 + b1) (bf16) ; xA = xB + h1 @ W2 + b2 (fp32)
        dim3 g1(FF / 64, S / 64, 1);
        gemm_bf16_kernel<1, true, true, false><<<g1, 128, 0, stream>>>(
            xn_bf, W1_bf, h1_bf, b1 + (size_t)l * FF, nullptr, S, FF, E, E, FF, FF, 1.0f, 0, 0, 0);
        dim3 g2(E / 64, S / 64, 1);
        gemm_bf16_kernel<0, true, false, true><<<g2, 128, 0, stream>>>(
            h1_bf, W2_bf, xA, b2 + (size_t)l * E, xB, S, E, FF, FF, E, E, 1.0f, 0, 0, 0);
    }

    // final activations -> d_out[0 : S*E]
    hipMemcpyAsync(d_out, xA, (size_t)S * E * sizeof(float), hipMemcpyDeviceToDevice, stream);
}
